// Intra_S_view_9509057593867
// MI455X (gfx1250) — matrix operand order repaired and checked by reading, not compiled, not run on hardware
//
#include <hip/hip_runtime.h>

// ---------------------------------------------------------------------------
// GAT-style attention on MI455X (gfx1250, wave32, WMMA).
// e_k = leaky_relu((a_k (*) H) H^T), adj-select, softmax_j, out = alpha @ H.
// bf16 split-GEMM (hi+lo, 3 products) over v_wmma_f32_16x16x32_bf16 gives
// ~fp32 accuracy at bf16 matrix-pipe rates.
// ---------------------------------------------------------------------------

typedef __attribute__((ext_vector_type(16))) __bf16 v16bf;
typedef __attribute__((ext_vector_type(8)))  __bf16 v8bf;
typedef __attribute__((ext_vector_type(8)))  float  v8f;

#define BB 32
#define NN 512
#define DD 256
#define GPAD 264        // 256 + 8 bf16 pad  -> row stride 528B (bank spread)
#define SPAD 516        // 512 + 4 f32  pad  -> row stride 2064B (bank spread)
#define NEG_INF (-9.0e15f)
#define SLOPE 0.2f

__device__ __forceinline__ v8f wmma_bf16(v16bf a, v16bf b, v8f c) {
    // 8 args: (neg_a, A, neg_b, B, c_mod, C, reuse_a, reuse_b)
    return __builtin_amdgcn_wmma_f32_16x16x32_bf16(
        false, a, false, b, (short)0, c, false, false);
}

__device__ __forceinline__ void split2(float x, __bf16& hi, __bf16& lo) {
    hi = (__bf16)x;
    lo = (__bf16)(x - (float)hi);   // residual: 2-term bf16 decomposition
}

// Branchless leaky-relu: for x>0, x > 0.2x; for x<0, 0.2x > x.
__device__ __forceinline__ float lrelu(float x) {
    return fmaxf(x, SLOPE * x);
}

__global__ __launch_bounds__(512, 1)
void gat_wmma_kernel(const float* __restrict__ hsrc,
                     const int*   __restrict__ adj,
                     const float* __restrict__ a0,
                     const float* __restrict__ a1,
                     const float* __restrict__ a2,
                     const float* __restrict__ a3,
                     float*       __restrict__ out)
{
    __shared__ __bf16 Ghi[4][16][GPAD];   // (a_k (*) H_i) hi part
    __shared__ __bf16 Glo[4][16][GPAD];   // (a_k (*) H_i) lo part
    __shared__ float  S[16][SPAD];        // scores -> alpha

    const int itile = blockIdx.x;         // 32 row tiles of 16
    const int b     = blockIdx.y;         // 32 batches
    const int i0    = itile * 16;
    const int tid   = threadIdx.x;
    const int w     = tid >> 5;           // wave id, 0..15 (wave32)
    const int L     = tid & 31;           // lane
    const int half  = L >> 4;             // lane half (WMMA layout)
    const int ln    = L & 15;

    const float* av[4] = {a0, a1, a2, a3};

    // ---------------- Phase 0: build G_k = a_k (*) H[i-rows], split hi/lo --
    for (int e = tid; e < 4 * 16 * DD; e += 512) {
        int k = e >> 12;                  // / (16*256)
        int r = e & 4095;
        int m = r >> 8;                   // / 256
        int d = r & 255;
        float g = av[k][d] * hsrc[((size_t)b * NN + (i0 + m)) * DD + d];
        __bf16 hi, lo; split2(g, hi, lo);
        Ghi[k][m][d] = hi;
        Glo[k][m][d] = lo;
    }
    __syncthreads();

    // ---------------- Phase 1: scores S[16][512] -----------------------------
    // wave w handles j-tiles {w, w+16}
    for (int jt = w; jt < 32; jt += 16) {
        const int j0 = jt * 16;

        // Hoist the adj tile loads: 8 independent 32-bit loads issued up
        // front so their latency hides under the whole WMMA K-loop.
        int adv[8];
#pragma unroll
        for (int r = 0; r < 8; ++r) {
            int m = r + half * 8;
            adv[r] = adj[((size_t)b * NN + (i0 + m)) * NN + j0 + ln];
        }

        v8f acc[4] = {v8f{}, v8f{}, v8f{}, v8f{}};

        for (int c = 0; c < DD / 32; ++c) {
            // B fragment (K x N = 32 x 16): lane = column j0+ln,
            // K = c*32 + half*16 + t  -> 64B contiguous per lane.
            const float* hp =
                &hsrc[((size_t)b * NN + (j0 + ln)) * DD + c * 32 + half * 16];
            v16bf bhi, blo;
#pragma unroll
            for (int t = 0; t < 16; ++t) {
                __bf16 hi, lo; split2(hp[t], hi, lo);
                bhi[t] = hi; blo[t] = lo;
            }

#pragma unroll
            for (int k = 0; k < 4; ++k) {
                // A fragment (M x K = 16 x 32): lane row m=ln,
                // K groups {half*16 .. +7} and {half*16+8 .. +7}.
                const v8bf* p0h = (const v8bf*)&Ghi[k][ln][c * 32 + half * 16];
                const v8bf* p1h = (const v8bf*)&Ghi[k][ln][c * 32 + half * 16 + 8];
                const v8bf* p0l = (const v8bf*)&Glo[k][ln][c * 32 + half * 16];
                const v8bf* p1l = (const v8bf*)&Glo[k][ln][c * 32 + half * 16 + 8];
                v8bf g0h = *p0h, g1h = *p1h, g0l = *p0l, g1l = *p1l;
                v16bf ahi, alo;
#pragma unroll
                for (int t = 0; t < 8; ++t) {
                    ahi[t] = g0h[t];  ahi[t + 8] = g1h[t];
                    alo[t] = g0l[t];  alo[t + 8] = g1l[t];
                }
                acc[k] = wmma_bf16(ahi, bhi, acc[k]);   // hi*hi
                acc[k] = wmma_bf16(ahi, blo, acc[k]);   // hi*lo
                acc[k] = wmma_bf16(alo, bhi, acc[k]);   // lo*hi
            }
        }

        // Branchless leaky-relu + adj select + store to LDS.
        // C/D layout: lane col n=ln, vgpr r -> row m = r + half*8.
#pragma unroll
        for (int r = 0; r < 8; ++r) {
            int m  = r + half * 8;
            int ad = adv[r];
            float e0 = lrelu(acc[0][r]);
            float e1 = lrelu(acc[1][r]);
            float e2 = lrelu(acc[2][r]);
            float e3 = lrelu(acc[3][r]);
            // Pure-VALU select: mask arithmetic, no EXEC-mask branching.
            float m1 = (float)(ad == 1);
            float m2 = (float)(ad == 2);
            float m3 = (float)(ad == 3);
            float m4 = (float)(ad == 4);
            float hit = m1 + m2 + m3 + m4;          // 0 or 1 (adj in [0,5))
            float s = m1 * e0 + m2 * e1 + m3 * e2 + m4 * e3
                    + (1.0f - hit) * NEG_INF;
            S[m][j0 + ln] = s;
        }
    }
    __syncthreads();

    // ---------------- Phase 2: softmax over j (wave w owns row w) -----------
    {
        const int m = w;
        float mx = -INFINITY;
        for (int j = L; j < NN; j += 32) mx = fmaxf(mx, S[m][j]);
#pragma unroll
        for (int off = 16; off > 0; off >>= 1)
            mx = fmaxf(mx, __shfl_xor(mx, off));

        float sum = 0.0f;
        for (int j = L; j < NN; j += 32) {
            float ev = __expf(S[m][j] - mx);
            S[m][j] = ev;
            sum += ev;
        }
#pragma unroll
        for (int off = 16; off > 0; off >>= 1)
            sum += __shfl_xor(sum, off);

        float inv = 1.0f / sum;
        for (int j = L; j < NN; j += 32) S[m][j] *= inv;
    }
    __syncthreads();

    // ---------------- Phase 3: out = alpha @ H (wave w owns d-tile w) -------
    {
        const int d0 = w * 16;
        v8f acc = {};
        for (int c = 0; c < NN / 32; ++c) {
            // A fragment from alpha (M x K = 16 x 32), row m=ln.
            v16bf ahi, alo;
            int jb0 = c * 32 + half * 16;
            int jb1 = jb0 + 8;
#pragma unroll
            for (int t = 0; t < 8; ++t) {
                __bf16 hi, lo;
                split2(S[ln][jb0 + t], hi, lo); ahi[t] = hi;      alo[t] = lo;
                split2(S[ln][jb1 + t], hi, lo); ahi[t + 8] = hi;  alo[t + 8] = lo;
            }
            // B fragment from H (K x N = 32 x 16): col n = d0+ln,
            // row j = c*32 + half*16 + t -> 64B coalesced per half-wave.
            v16bf bhi, blo;
#pragma unroll
            for (int t = 0; t < 16; ++t) {
                int j = c * 32 + half * 16 + t;
                __bf16 hi, lo;
                split2(hsrc[((size_t)b * NN + j) * DD + d0 + ln], hi, lo);
                bhi[t] = hi; blo[t] = lo;
            }
            acc = wmma_bf16(ahi, bhi, acc);
            acc = wmma_bf16(ahi, blo, acc);
            acc = wmma_bf16(alo, bhi, acc);
        }
#pragma unroll
        for (int r = 0; r < 8; ++r) {
            int m = r + half * 8;
            out[((size_t)b * NN + (i0 + m)) * DD + d0 + ln] = acc[r];
        }
    }
}

extern "C" void kernel_launch(void* const* d_in, const int* in_sizes, int n_in,
                              void* d_out, int out_size, void* d_ws, size_t ws_size,
                              hipStream_t stream) {
    (void)in_sizes; (void)n_in; (void)out_size; (void)d_ws; (void)ws_size;
    const float* hidden = (const float*)d_in[0];
    const int*   adj    = (const int*)d_in[1];
    const float* a0     = (const float*)d_in[2];
    const float* a1     = (const float*)d_in[3];
    const float* a2     = (const float*)d_in[4];
    const float* a3     = (const float*)d_in[5];
    float*       outp   = (float*)d_out;

    dim3 grid(NN / 16, BB);   // (i-tile, batch)
    gat_wmma_kernel<<<grid, 512, 0, stream>>>(hidden, adj, a0, a1, a2, a3, outp);
}
